// LinearAttention_6339371729053
// MI455X (gfx1250) — matrix operand order repaired and checked by reading, not compiled, not run on hardware
//
#include <hip/hip_runtime.h>
#include <hip/hip_bf16.h>

// Linear attention (causal, elu+1 feature map) for MI455X / gfx1250.
// B=2, L=1024, D=1024, H=16, d=64.
//
// v3: all operands pre-converted to bf16 once; GEMM/attention hot loops do
// pure b128 loads + v_wmma_f32_16x16x32_bf16 (no conversions in-loop).
//
//   0. cvt_bf16: query/key/value + Wq/Wk/Wv/Wo  fp32 -> bf16      (7 calls)
//   1. gemm_wmma: q' = phi(scale*query@Wq^T) -> bf16 [B,H,L,d]    (mode 1)
//                 k' = phi(scale*key  @Wk^T) -> bf16 [B,H,L,d]    (mode 1)
//                 v  =            value@Wv^T -> bf16 [B,H,L,d]    (mode 2)
//   2. linattn_wmma: chunked causal scan (16-row chunks, f32 state S in LDS)
//                 -> bf16 att [B,L,D]
//   3. gemm_wmma: out = att@Wo^T -> fp32 d_out                    (mode 0)

#define HEADS  16
#define DMODEL 1024
#define DHEAD  64
#define LSEQ   1024
#define NBATCH 2
#define EPSV   1e-6f

typedef __bf16 bf16_t;
typedef bf16_t v4bf  __attribute__((ext_vector_type(4)));
typedef bf16_t v8bf  __attribute__((ext_vector_type(8)));
typedef bf16_t v16bf __attribute__((ext_vector_type(16)));
typedef float  v8f   __attribute__((ext_vector_type(8)));

// ---------------------------------------------------------------------------
// bf16 fragment loaders (contiguous rows; 16B-aligned).
//
// A-matrix 16x32 bf16 (ISA 7.12.2): lane = row m (lane&15), hi = lane>>4.
//   VGPR v 0..7: K = 16*hi + 2v + {0,1} (code's own order, not ISA's) -> 16 contiguous bf16 at k0+16*hi
static __device__ __forceinline__ v16bf afrag_bf(const bf16_t* Arow, int k0,
                                                 int hi) {
  v8bf lo = *(const v8bf*)(Arow + k0 + 16 * hi);
  v8bf up = *(const v8bf*)(Arow + k0 + 16 * hi + 8);
  return __builtin_shufflevector(lo, up, 0, 1, 2, 3, 4, 5, 6, 7, 8, 9, 10, 11,
                                 12, 13, 14, 15);
}

// B-matrix 32x16 bf16: lane = column n (lane&15), hi = lane>>4.
//   VGPR v 0..7: K = 16*hi + 2v + {0,1} -> 16 contiguous bf16 of B column n.
//   For B[k][n] = W[n][k] (row-major W): 16 contiguous bf16 of row n @ k0+16hi.
static __device__ __forceinline__ v16bf bfrag_bf(const bf16_t* Wrow, int k0,
                                                 int hi) {
  v8bf lo = *(const v8bf*)(Wrow + k0 + 16 * hi);
  v8bf up = *(const v8bf*)(Wrow + k0 + 16 * hi + 8);
  return __builtin_shufflevector(lo, up, 0, 1, 2, 3, 4, 5, 6, 7, 8, 9, 10, 11,
                                 12, 13, 14, 15);
}

// Strided / zero-padded variants (LDS tiles; kmax/kstride compile-time).
static __device__ __forceinline__ v16bf afrag_bf_str(const bf16_t* base,
                                                     int kstride, int kmax,
                                                     int hi) {
  v16bf a;
#pragma unroll
  for (int v = 0; v < 8; ++v) {
    int k = 16 * hi + 2 * v;
    a[2 * v]     = (k     < kmax) ? base[k * kstride]       : (bf16_t)0.f;
    a[2 * v + 1] = (k + 1 < kmax) ? base[(k + 1) * kstride] : (bf16_t)0.f;
  }
  return a;
}

static __device__ __forceinline__ v16bf bfrag_bf_str(const bf16_t* base,
                                                     int kstride, int kmax,
                                                     int hi) {
  v16bf b;
#pragma unroll
  for (int v = 0; v < 8; ++v) {
    int k = 16 * hi + 2 * v;
    b[2 * v]     = (k     < kmax) ? base[k * kstride]       : (bf16_t)0.f;
    b[2 * v + 1] = (k + 1 < kmax) ? base[(k + 1) * kstride] : (bf16_t)0.f;
  }
  return b;
}

// f32-source strided B fragment (for the f32 state S; cvt only here).
static __device__ __forceinline__ v16bf bfrag_f32_str(const float* base,
                                                      int kstride, int k0,
                                                      int hi) {
  v16bf b;
#pragma unroll
  for (int v = 0; v < 8; ++v) {
    int k = k0 + 16 * hi + 2 * v;
    b[2 * v]     = (bf16_t)base[k * kstride];
    b[2 * v + 1] = (bf16_t)base[(k + 1) * kstride];
  }
  return b;
}

// ---------------------------------------------------------------------------
// Elementwise fp32 -> bf16 (vectorized: float4 in, 4x bf16 out).
__global__ __launch_bounds__(256) void cvt_bf16(const float* __restrict__ in,
                                                bf16_t* __restrict__ out,
                                                int n4) {
  int i = blockIdx.x * 256 + threadIdx.x;
  if (i < n4) {
    float4 x = ((const float4*)in)[i];
    v4bf o;
    o[0] = (bf16_t)x.x; o[1] = (bf16_t)x.y; o[2] = (bf16_t)x.z; o[3] = (bf16_t)x.w;
    ((v4bf*)out)[i] = o;
  }
}

// ---------------------------------------------------------------------------
// C = act(scale * A @ W^T).  A:[M,1024] bf16, W:[N,1024] bf16.
// mode 0: store fp32 [M,N]                      -- output projection
// mode 1: phi(scale*x), bf16 -> [B,H,L,d]       -- q, k
// mode 2: plain, bf16 -> [B,H,L,d]              -- v
// Block 256 (8 waves); wave w owns tile (blockIdx.y*128 + w*16, blockIdx.x*16).
__global__ __launch_bounds__(256) void gemm_wmma(
    const bf16_t* __restrict__ A, const bf16_t* __restrict__ W,
    float* __restrict__ outf, bf16_t* __restrict__ outb, int M, int N,
    int mode, float scale) {
  const int lane = threadIdx.x & 31;
  const int wave = threadIdx.x >> 5;
  const int hi = lane >> 4;
  const int ln = lane & 15;
  const int mbase = blockIdx.y * 128 + wave * 16;
  const int nbase = blockIdx.x * 16;

  const bf16_t* Arow = A + (size_t)(mbase + ln) * DMODEL;
  const bf16_t* Wrow = W + (size_t)(nbase + ln) * DMODEL;

  v8f acc = {};
#pragma unroll 4
  for (int k0 = 0; k0 < DMODEL; k0 += 32) {
    __builtin_prefetch(Arow + k0 + 256, 0, 3);  // global_prefetch_b8
    __builtin_prefetch(Wrow + k0 + 256, 0, 3);
    v16bf af = afrag_bf(Arow, k0, hi);
    v16bf bf = bfrag_bf(Wrow, k0, hi);
    acc = __builtin_amdgcn_wmma_f32_16x16x32_bf16(false, af, false, bf,
                                                  (short)0, acc, false, false);
  }

#pragma unroll
  for (int r = 0; r < 8; ++r) {
    int mm = mbase + r + 8 * hi;  // C/D layout: VGPR r, lanes>=16 -> m+8
    int nn = nbase + ln;
    float vvv = acc[r];
    if (mode == 0) {
      outf[(size_t)mm * N + nn] = vvv;
    } else {
      if (mode == 1) {
        vvv *= scale;
        vvv = (vvv > 0.f) ? (vvv + 1.f) : __expf(vvv);  // elu(x)+1
      }
      int bb = mm / LSEQ, ll = mm % LSEQ;
      int hh = nn / DHEAD, dd = nn % DHEAD;
      outb[(((size_t)bb * HEADS + hh) * LSEQ + ll) * DHEAD + dd] = (bf16_t)vvv;
    }
  }
}

// ---------------------------------------------------------------------------
// Chunked causal linear-attention scan.  One workgroup (4 waves) per (b,h).
// Per 16-row chunk:
//   P   = tril(Qc @ Kc^T)                      (1 tile,  2 WMMA, wave 0)
//   den = Qc @ z + rowsum(P) + eps             (scalar, 16 threads)
//   O   = (Qc @ S + P @ Vc) / den              (4 tiles, 3 WMMA each)
//   S  += Kc^T @ Vc ; z += colsum(Kc)          (16 tiles, 1 WMMA each)
__global__ __launch_bounds__(128) void linattn_wmma(
    const bf16_t* __restrict__ QP, const bf16_t* __restrict__ KP,
    const bf16_t* __restrict__ V, bf16_t* __restrict__ att) {
  __shared__ float S[DHEAD * DHEAD];                 // f32 running state
  __shared__ float z[DHEAD];                         // f32 running k' sum
  __shared__ __align__(16) bf16_t Qc[16 * DHEAD];
  __shared__ __align__(16) bf16_t Kc[16 * DHEAD];
  __shared__ __align__(16) bf16_t Vc[16 * DHEAD];
  __shared__ float Pmf[16 * 16];                     // masked P, f32 (den)
  __shared__ __align__(16) bf16_t Pmb[16 * 16];      // masked P, bf16 (frag)
  __shared__ float denv[16];

  const int bh = blockIdx.x;
  const int b = bh / HEADS, h = bh % HEADS;
  const size_t base = (size_t)bh * LSEQ * DHEAD;
  const int tid = threadIdx.x;
  const int lane = tid & 31, wave = tid >> 5;
  const int hi = lane >> 4, ln = lane & 15;

  for (int i = tid; i < DHEAD * DHEAD; i += 128) S[i] = 0.f;
  if (tid < DHEAD) z[tid] = 0.f;
  __syncthreads();

  for (int t = 0; t < LSEQ / 16; ++t) {
    const size_t coff = base + (size_t)t * 16 * DHEAD;
    // stage chunk: 16x64 bf16 = 2 KB per matrix = one uint4 per thread
    ((uint4*)Qc)[tid] = ((const uint4*)(QP + coff))[tid];
    ((uint4*)Kc)[tid] = ((const uint4*)(KP + coff))[tid];
    ((uint4*)Vc)[tid] = ((const uint4*)(V + coff))[tid];
    __syncthreads();

    if (wave == 0) {  // wave-uniform branch: EXEC all-ones for WMMA
      v8f p = {};
#pragma unroll
      for (int k0 = 0; k0 < DHEAD; k0 += 32) {
        v16bf aq = afrag_bf(&Qc[ln * DHEAD], k0, hi);
        v16bf bk = bfrag_bf(&Kc[ln * DHEAD], k0, hi);  // B[k][n]=Kc[n][k]
        p = __builtin_amdgcn_wmma_f32_16x16x32_bf16(false, aq, false, bk,
                                                    (short)0, p, false, false);
      }
#pragma unroll
      for (int r = 0; r < 8; ++r) {
        int mm = r + 8 * hi, nn = ln;
        float pv = (nn <= mm) ? p[r] : 0.f;  // causal mask (diag kept)
        Pmf[mm * 16 + nn] = pv;
        Pmb[mm * 16 + nn] = (bf16_t)pv;
      }
    }
    __syncthreads();

    if (tid < 16) {  // den_m = q_m . z_prev + sum_{n<=m} P[m][n] + eps
      float s = EPSV;
      for (int k = 0; k < DHEAD; ++k) s += (float)Qc[tid * DHEAD + k] * z[k];
      for (int nn = 0; nn <= tid; ++nn) s += Pmf[tid * 16 + nn];
      denv[tid] = s;
    }
    __syncthreads();

    {  // wave j -> output columns [16j, 16j+16)
      const int j = wave;
      v8f acc = {};
#pragma unroll
      for (int k0 = 0; k0 < DHEAD; k0 += 32) {
        v16bf aq = afrag_bf(&Qc[ln * DHEAD], k0, hi);
        v16bf bs = bfrag_f32_str(&S[16 * j + ln], DHEAD, k0, hi);
        acc = __builtin_amdgcn_wmma_f32_16x16x32_bf16(false, aq, false, bs,
                                                      (short)0, acc, false, false);
      }
      v16bf ap = afrag_bf_str(&Pmb[ln * 16], 1, 16, hi);        // K=16 pad
      v16bf bv = bfrag_bf_str(&Vc[16 * j + ln], DHEAD, 16, hi);
      acc = __builtin_amdgcn_wmma_f32_16x16x32_bf16(false, ap, false, bv,
                                                    (short)0, acc, false, false);
#pragma unroll
      for (int r = 0; r < 8; ++r) {
        int mm = r + 8 * hi;
        int ll = t * 16 + mm;
        int col = h * DHEAD + 16 * j + ln;
        att[((size_t)b * LSEQ + ll) * DMODEL + col] =
            (bf16_t)(acc[r] / denv[mm]);
      }
    }
    __syncthreads();  // all S reads complete before the state update

    {  // S[16*wave .. +16][:] += Kc^T @ Vc   (A[m][kt] = Kc[kt][16*wave+m])
      const int mi = wave;
      v16bf ak = afrag_bf_str(&Kc[16 * mi + ln], DHEAD, 16, hi);
#pragma unroll
      for (int jj = 0; jj < 4; ++jj) {
        v16bf bv = bfrag_bf_str(&Vc[16 * jj + ln], DHEAD, 16, hi);
        v8f d = {};
        d = __builtin_amdgcn_wmma_f32_16x16x32_bf16(false, ak, false, bv,
                                                    (short)0, d, false, false);
#pragma unroll
        for (int r = 0; r < 8; ++r) {
          int mm = 16 * mi + r + 8 * hi;
          int nn = 16 * jj + ln;
          S[mm * DHEAD + nn] += d[r];  // each (mm,nn) owned by one lane
        }
      }
    }
    if (tid < DHEAD) {  // z += colsum(Kc)
      float s = 0.f;
      for (int mrow = 0; mrow < 16; ++mrow)
        s += (float)Kc[mrow * DHEAD + tid];
      z[tid] += s;
    }
    __syncthreads();
  }
}

extern "C" void kernel_launch(void* const* d_in, const int* in_sizes, int n_in,
                              void* d_out, int out_size, void* d_ws, size_t ws_size,
                              hipStream_t stream) {
  const float* query = (const float*)d_in[0];
  const float* key   = (const float*)d_in[1];
  const float* value = (const float*)d_in[2];
  const float* Wq    = (const float*)d_in[3];
  const float* Wk    = (const float*)d_in[4];
  const float* Wv    = (const float*)d_in[5];
  const float* Wo    = (const float*)d_in[6];
  float* out = (float*)d_out;

  const int M = NBATCH * LSEQ;          // 2048
  const int N = DMODEL;                 // 1024
  const size_t MK = (size_t)M * DMODEL; // 2 Mi elements
  const size_t KK = (size_t)DMODEL * DMODEL;

  // Workspace: all bf16 -> 36 MB total.
  bf16_t* p = (bf16_t*)d_ws;
  bf16_t* Xq  = p; p += MK;   // query bf16
  bf16_t* Xk  = p; p += MK;
  bf16_t* Xv  = p; p += MK;
  bf16_t* Wqb = p; p += KK;
  bf16_t* Wkb = p; p += KK;
  bf16_t* Wvb = p; p += KK;
  bf16_t* Wob = p; p += KK;
  bf16_t* qp  = p; p += MK;   // [B,H,L,d]
  bf16_t* kp  = p; p += MK;
  bf16_t* vv  = p; p += MK;
  bf16_t* att = p; p += MK;   // [B,L,D]

  const float scale = 0.125f;  // d^-0.5 = 64^-0.5

  // 0) fp32 -> bf16 conversions (bandwidth-trivial, one pass)
  {
    int n4a = (int)(MK / 4), n4w = (int)(KK / 4);
    cvt_bf16<<<dim3(n4a / 256), dim3(256), 0, stream>>>(query, Xq, n4a);
    cvt_bf16<<<dim3(n4a / 256), dim3(256), 0, stream>>>(key,   Xk, n4a);
    cvt_bf16<<<dim3(n4a / 256), dim3(256), 0, stream>>>(value, Xv, n4a);
    cvt_bf16<<<dim3(n4w / 256), dim3(256), 0, stream>>>(Wq, Wqb, n4w);
    cvt_bf16<<<dim3(n4w / 256), dim3(256), 0, stream>>>(Wk, Wkb, n4w);
    cvt_bf16<<<dim3(n4w / 256), dim3(256), 0, stream>>>(Wv, Wvb, n4w);
    cvt_bf16<<<dim3(n4w / 256), dim3(256), 0, stream>>>(Wo, Wob, n4w);
  }

  dim3 blk(256);
  dim3 grd(N / 16, M / 128);
  gemm_wmma<<<grd, blk, 0, stream>>>(Xq, Wqb, nullptr, qp, M, N, 1, scale);
  gemm_wmma<<<grd, blk, 0, stream>>>(Xk, Wkb, nullptr, kp, M, N, 1, scale);
  gemm_wmma<<<grd, blk, 0, stream>>>(Xv, Wvb, nullptr, vv, M, N, 2, 1.0f);

  linattn_wmma<<<dim3(NBATCH * HEADS), dim3(128), 0, stream>>>(qp, kp, vv, att);

  gemm_wmma<<<grd, blk, 0, stream>>>(att, Wob, out, nullptr, M, N, 0, 1.0f);
}
